// GNNEncoder_3624952398185
// MI455X (gfx1250) — matrix operand order repaired and checked by reading, not compiled, not run on hardware
//
#include <hip/hip_runtime.h>
#include <hip/hip_bf16.h>

typedef _Float16 half_t;
typedef __attribute__((ext_vector_type(16))) _Float16 v16h;
typedef __attribute__((ext_vector_type(8)))  float    v8f;

#define NN 100000
#define EE 1600000
#define HD 64
#define GG 128
#define ZZ 32

// ---------------------------------------------------------------------------
// Generic zero kernel (workspace init; graph-capture safe)
// ---------------------------------------------------------------------------
__global__ void zero_kernel(float* __restrict__ p, int n) {
    int t = blockIdx.x * blockDim.x + threadIdx.x;
    if (t < n) p[t] = 0.0f;
}

// ---------------------------------------------------------------------------
// Edge scatter: agg[dst[e]][:] += h[src[e]][:]
// 16 threads per edge, each thread handles 4 contiguous dims (float4 gather,
// 4x global_atomic_add_f32 scatter). Consecutive threads cover one 256B row.
// ---------------------------------------------------------------------------
__global__ void scatter_kernel(const float* __restrict__ h,
                               const int* __restrict__ src,
                               const int* __restrict__ dst,
                               float* __restrict__ agg, int E_) {
    int t = blockIdx.x * blockDim.x + threadIdx.x;
    if (t >= E_ * 16) return;
    int e  = t >> 4;
    int c4 = (t & 15) * 4;
    int s = src[e];
    int d = dst[e];
    const float4 v = *(const float4*)(h + (size_t)s * HD + c4);
    float* a = agg + (size_t)d * HD + c4;
    atomicAdd(a + 0, v.x);
    atomicAdd(a + 1, v.y);
    atomicAdd(a + 2, v.z);
    atomicAdd(a + 3, v.w);
}

__device__ __forceinline__ void ld8(const float* p, float* out) {
    float4 a = *(const float4*)p;
    float4 b = *(const float4*)(p + 4);
    out[0] = a.x; out[1] = a.y; out[2] = a.z; out[3] = a.w;
    out[4] = b.x; out[5] = b.y; out[6] = b.z; out[7] = b.w;
}

// ---------------------------------------------------------------------------
// GIN MLP via WMMA: h2 = relu(((1+eps)*h + agg) @ W1 + b1) @ W2 + b2
// One wave per 16-row tile, 8 waves (256 threads) per block -> 128 rows per
// block. Weights are staged once per block into LDS, pre-swizzled into WMMA
// B-fragment order so each fragment is one contiguous 32B (2x ds_load_b128)
// read. ReLU(GEMM1) output is written to LDS directly in A-fragment order,
// so GEMM2's A fragments are contiguous v16h loads too. No loops carry WMMA
// operands -> bounded register lifetimes, no scratch spills.
// NOTE: h2_out may alias agg (same-row read-before-write within one wave),
// so those two pointers carry no __restrict__.
// ---------------------------------------------------------------------------
__global__ void __launch_bounds__(256)
gin_mlp_kernel(const float* __restrict__ h_in,
               const float* agg,
               const float* __restrict__ eps_p,
               const float* __restrict__ W1g, const float* __restrict__ b1g,
               const float* __restrict__ W2g, const float* __restrict__ b2g,
               float* h2_out,
               float* __restrict__ gsum, float* __restrict__ gsumsq,
               int N_) {
    // Fragment-swizzled f16 weights: index ((nt*2+kt)*32 + lane)*16 + j
    __shared__ __attribute__((aligned(32))) half_t w1s[8 * 32 * 16];
    __shared__ __attribute__((aligned(32))) half_t w2s[8 * 32 * 16];
    // Per-wave Z staging, already in A-fragment order: (kt*32 + lane)*16 + j
    __shared__ __attribute__((aligned(32))) half_t zsw[8][2 * 32 * 16];
    __shared__ float ssum[HD];
    __shared__ float ssq[HD];

    const int tid = threadIdx.x;
    // Stage + swizzle weights (coalesced global read, scattered LDS write).
    for (int i = tid; i < HD * HD; i += 256) {
        int k = i >> 6, n = i & 63;
        int dstIdx = (((n >> 4) * 2 + (k >> 5)) * 32 + ((k >> 3) & 1) * 16 + (n & 15)) * 16
                     + (k & 7) + 8 * ((k >> 4) & 1);
        w1s[dstIdx] = (half_t)W1g[i];
        w2s[dstIdx] = (half_t)W2g[i];
    }
    if (tid < HD) { ssum[tid] = 0.0f; ssq[tid] = 0.0f; }
    __syncthreads();

    const int wave = tid >> 5;
    const int lane = tid & 31;
    const int m    = lane & 15;          // A row / B-C column within tile
    const int hi   = lane >> 4;          // lane half select
    const int hi2  = (m >> 3) & 1;       // col-half bit for swizzled Z writes
    const int row0 = (blockIdx.x * 8 + wave) * 16;
    const bool active = (row0 < N_);     // N_ % 16 == 0 -> full tiles only

    float accS[4] = {0.f, 0.f, 0.f, 0.f};
    float accQ[4] = {0.f, 0.f, 0.f, 0.f};

    if (active) {
        const float epsv = 1.0f + eps_p[0];
        // ---- A fragments for GEMM1: x = (1+eps)*h + agg, f32 -> f16 -------
        const float* hrow = h_in + (size_t)(row0 + m) * HD;
        const float* arow = agg  + (size_t)(row0 + m) * HD;
        v16h afrag[2];
        for (int kt = 0; kt < 2; ++kt) {
            const int cb = 32 * kt + 8 * hi;
            float hv[16], av[16];
            ld8(hrow + cb,      hv);     // cols cb..cb+7   -> j = 0..7
            ld8(hrow + cb + 16, hv + 8); // cols cb+16..+23 -> j = 8..15
            ld8(arow + cb,      av);
            ld8(arow + cb + 16, av + 8);
#pragma unroll
            for (int j = 0; j < 16; ++j)
                afrag[kt][j] = (half_t)(epsv * hv[j] + av[j]);
        }

        // ---- GEMM1 + bias + ReLU -> LDS in A-fragment order ---------------
        for (int nt = 0; nt < 4; ++nt) {
            const v16h bf0 = *(const v16h*)&w1s[((nt * 2 + 0) * 32 + lane) * 16];
            const v16h bf1 = *(const v16h*)&w1s[((nt * 2 + 1) * 32 + lane) * 16];
            v8f c = {};
            c = __builtin_amdgcn_wmma_f32_16x16x32_f16(false, afrag[0], false, bf0,
                                                       (short)0, c, false, false);
            c = __builtin_amdgcn_wmma_f32_16x16x32_f16(false, afrag[1], false, bf1,
                                                       (short)0, c, false, false);
            const float bias = b1g[16 * nt + m];
            // destination (A-fragment order): lane' = r + 8*hi + 16*hi2,
            // kt' = nt>>1, j' = (m&7) + 8*(nt&1)
            half_t* zw = &zsw[wave][((nt >> 1) * 32 + 8 * hi + 16 * hi2) * 16
                                    + (m & 7) + 8 * (nt & 1)];
#pragma unroll
            for (int r = 0; r < 8; ++r) {
                float v = c[r] + bias;
                v = v > 0.0f ? v : 0.0f;
                zw[r * 16] = (half_t)v;
            }
        }

        // ---- A fragments for GEMM2: contiguous v16h loads -----------------
        v16h zfrag[2];
        zfrag[0] = *(const v16h*)&zsw[wave][(0 * 32 + lane) * 16];
        zfrag[1] = *(const v16h*)&zsw[wave][(1 * 32 + lane) * 16];

        // ---- GEMM2 + bias -> h2 (f32) + BN partial stats ------------------
        for (int nt = 0; nt < 4; ++nt) {
            const v16h bf0 = *(const v16h*)&w2s[((nt * 2 + 0) * 32 + lane) * 16];
            const v16h bf1 = *(const v16h*)&w2s[((nt * 2 + 1) * 32 + lane) * 16];
            v8f c = {};
            c = __builtin_amdgcn_wmma_f32_16x16x32_f16(false, zfrag[0], false, bf0,
                                                       (short)0, c, false, false);
            c = __builtin_amdgcn_wmma_f32_16x16x32_f16(false, zfrag[1], false, bf1,
                                                       (short)0, c, false, false);
            const float bias = b2g[16 * nt + m];
            float ls = 0.0f, lq = 0.0f;
#pragma unroll
            for (int r = 0; r < 8; ++r) {
                const float v = c[r] + bias;
                h2_out[(size_t)(row0 + r + 8 * hi) * HD + (16 * nt + m)] = v;
                ls += v;
                lq += v * v;
            }
            accS[nt] += ls;
            accQ[nt] += lq;
        }
    }

    // ---- block-level BN stat reduction ------------------------------------
    if (active) {
#pragma unroll
        for (int nt = 0; nt < 4; ++nt) {
            atomicAdd(&ssum[16 * nt + m], accS[nt]);
            atomicAdd(&ssq [16 * nt + m], accQ[nt]);
        }
    }
    __syncthreads();
    if (tid < HD)          atomicAdd(&gsum[tid],        ssum[tid]);
    else if (tid < 2 * HD) atomicAdd(&gsumsq[tid - HD], ssq[tid - HD]);
}

// ---------------------------------------------------------------------------
// BN statistics finalize (biased variance, as torch BN in training mode)
// ---------------------------------------------------------------------------
__global__ void bnstats_kernel(const float* __restrict__ gsum,
                               const float* __restrict__ gsumsq,
                               float* __restrict__ mu, float* __restrict__ rsig,
                               float invN) {
    int c = threadIdx.x;
    float m = gsum[c] * invN;
    float v = gsumsq[c] * invN - m * m;
    mu[c]   = m;
    rsig[c] = rsqrtf(v + 1e-5f);
}

// ---------------------------------------------------------------------------
// BN apply + ReLU (float4 elementwise)
// ---------------------------------------------------------------------------
__global__ void bnrelu_kernel(const float* __restrict__ h2,
                              const float* __restrict__ mu,
                              const float* __restrict__ rsig,
                              const float* __restrict__ gamma,
                              const float* __restrict__ beta,
                              float* __restrict__ hout, int N_) {
    int t = blockIdx.x * blockDim.x + threadIdx.x;   // N*16 quads
    if (t >= N_ * 16) return;
    int c = (t & 15) * 4;
    float4 v = *(const float4*)(h2 + (size_t)t * 4);
    float4 o;
    o.x = fmaxf((v.x - mu[c + 0]) * rsig[c + 0] * gamma[c + 0] + beta[c + 0], 0.0f);
    o.y = fmaxf((v.y - mu[c + 1]) * rsig[c + 1] * gamma[c + 1] + beta[c + 1], 0.0f);
    o.z = fmaxf((v.z - mu[c + 2]) * rsig[c + 2] * gamma[c + 2] + beta[c + 2], 0.0f);
    o.w = fmaxf((v.w - mu[c + 3]) * rsig[c + 3] * gamma[c + 3] + beta[c + 3], 0.0f);
    *(float4*)(hout + (size_t)t * 4) = o;
}

// ---------------------------------------------------------------------------
// Per-graph sum pooling + counts
// ---------------------------------------------------------------------------
__global__ void pool_kernel(const float* __restrict__ h,
                            const int* __restrict__ batch,
                            float* __restrict__ psum, float* __restrict__ pcnt,
                            int N_) {
    int t = blockIdx.x * blockDim.x + threadIdx.x;   // N*16 quads
    if (t >= N_ * 16) return;
    int i = t >> 4;
    int c = (t & 15) * 4;
    int g = batch[i];
    float4 v = *(const float4*)(h + (size_t)i * HD + c);
    float* p = psum + (size_t)g * HD + c;
    atomicAdd(p + 0, v.x);
    atomicAdd(p + 1, v.y);
    atomicAdd(p + 2, v.z);
    atomicAdd(p + 3, v.w);
    if ((t & 15) == 0) atomicAdd(&pcnt[g], 1.0f);
}

// ---------------------------------------------------------------------------
// Latent heads: out[0:G*Z) = mean, out[G*Z:2*G*Z) = log_var
// ---------------------------------------------------------------------------
__global__ void head_kernel(const float* __restrict__ psum,
                            const float* __restrict__ pcnt,
                            const float* __restrict__ Wm, const float* __restrict__ bm,
                            const float* __restrict__ Wv, const float* __restrict__ bv,
                            float* __restrict__ out) {
    int t = blockIdx.x * blockDim.x + threadIdx.x;   // 2*G*Z
    if (t >= 2 * GG * ZZ) return;
    int headv = t / (GG * ZZ);
    int g = (t / ZZ) % GG;
    int z = t & (ZZ - 1);
    const float* W = headv ? Wv : Wm;
    const float* b = headv ? bv : bm;
    float cnt = fmaxf(pcnt[g], 1.0f);
    float s = 0.0f;
    const float* hg = psum + (size_t)g * HD;
#pragma unroll 8
    for (int k = 0; k < HD; ++k) s += (hg[k] / cnt) * W[k * ZZ + z];
    out[t] = s + b[z];
}

// ---------------------------------------------------------------------------
// Host orchestration
// ---------------------------------------------------------------------------
extern "C" void kernel_launch(void* const* d_in, const int* in_sizes, int n_in,
                              void* d_out, int out_size, void* d_ws, size_t ws_size,
                              hipStream_t stream) {
    (void)n_in; (void)out_size; (void)ws_size;

    const float *x, *Wm, *bm, *Wv, *bv;
    const int *ei, *batch;
    const float *epsL[3], *W1L[3], *b1L[3], *W2L[3], *b2L[3], *gaL[3], *beL[3];

    if (in_sizes[0] == NN * HD) {
        // Insertion order: x, edge_index, batch, [eps,W1,b1,W2,b2,gamma,beta]x3, Wm,bm,Wv,bv
        x     = (const float*)d_in[0];
        ei    = (const int*)d_in[1];
        batch = (const int*)d_in[2];
        for (int l = 0; l < 3; ++l) {
            int base = 3 + 7 * l;
            epsL[l] = (const float*)d_in[base + 0];
            W1L[l]  = (const float*)d_in[base + 1];
            b1L[l]  = (const float*)d_in[base + 2];
            W2L[l]  = (const float*)d_in[base + 3];
            b2L[l]  = (const float*)d_in[base + 4];
            gaL[l]  = (const float*)d_in[base + 5];
            beL[l]  = (const float*)d_in[base + 6];
        }
        Wm = (const float*)d_in[24];
        bm = (const float*)d_in[25];
        Wv = (const float*)d_in[26];
        bv = (const float*)d_in[27];
    } else {
        // Sorted pytree order: batch, edge_index, params{Wm,Wv,bm,bv,
        //   layers[{W1,W2,b1,b2,beta,eps,gamma}]x3}, x
        batch = (const int*)d_in[0];
        ei    = (const int*)d_in[1];
        Wm = (const float*)d_in[2];
        Wv = (const float*)d_in[3];
        bm = (const float*)d_in[4];
        bv = (const float*)d_in[5];
        for (int l = 0; l < 3; ++l) {
            int base = 6 + 7 * l;
            W1L[l]  = (const float*)d_in[base + 0];
            W2L[l]  = (const float*)d_in[base + 1];
            b1L[l]  = (const float*)d_in[base + 2];
            b2L[l]  = (const float*)d_in[base + 3];
            beL[l]  = (const float*)d_in[base + 4];
            epsL[l] = (const float*)d_in[base + 5];
            gaL[l]  = (const float*)d_in[base + 6];
        }
        x = (const float*)d_in[27];
    }
    const int* src = ei;
    const int* dst = ei + EE;

    // Workspace layout (floats)
    float* hbuf   = (float*)d_ws;                 // N*64
    float* aggbuf = hbuf + (size_t)NN * HD;       // N*64 (also holds h2)
    float* gsum   = aggbuf + (size_t)NN * HD;     // 64
    float* gsumsq = gsum + HD;                    // 64
    float* mu     = gsumsq + HD;                  // 64
    float* rsig   = mu + HD;                      // 64
    float* psum   = rsig + HD;                    // G*64
    float* pcnt   = psum + GG * HD;               // G

    const int elemBlocks = (NN * 16 + 255) / 256;
    const int mlpBlocks  = (NN + 127) / 128;      // 8 waves x 16 rows per block
    const float* hcur = x;

    for (int l = 0; l < 3; ++l) {
        zero_kernel<<<(NN * HD + 255) / 256, 256, 0, stream>>>(aggbuf, NN * HD);
        zero_kernel<<<1, 128, 0, stream>>>(gsum, 2 * HD);
        scatter_kernel<<<(EE * 16 + 255) / 256, 256, 0, stream>>>(hcur, src, dst, aggbuf, EE);
        gin_mlp_kernel<<<mlpBlocks, 256, 0, stream>>>(
            hcur, aggbuf, epsL[l], W1L[l], b1L[l], W2L[l], b2L[l],
            aggbuf /* h2 aliases agg: same-row overwrite only */, gsum, gsumsq, NN);
        bnstats_kernel<<<1, HD, 0, stream>>>(gsum, gsumsq, mu, rsig, 1.0f / (float)NN);
        bnrelu_kernel<<<elemBlocks, 256, 0, stream>>>(aggbuf, mu, rsig, gaL[l], beL[l], hbuf, NN);
        hcur = hbuf;
    }

    zero_kernel<<<(GG * HD + GG + 255) / 256, 256, 0, stream>>>(psum, GG * HD + GG);
    pool_kernel<<<elemBlocks, 256, 0, stream>>>(hbuf, batch, psum, pcnt, NN);
    head_kernel<<<(2 * GG * ZZ + 255) / 256, 256, 0, stream>>>(
        psum, pcnt, Wm, bm, Wv, bv, (float*)d_out);
}
